// Qwen3_5VisionAttention_25168508354985
// MI455X (gfx1250) — matrix operand order repaired and checked by reading, not compiled, not run on hardware
//
#include <hip/hip_runtime.h>
#include <hip/hip_bf16.h>

// ---------------------------------------------------------------------------
// Qwen3.5 Vision Attention block for gfx1250 (MI455X), bf16 WMMA pipeline.
//   SEQ=4096, DIM=1280 (16 heads x 80), 8 segments of 512 (block-diag attn).
// GEMMs: LDS-tiled, double-buffered via GLOBAL_LOAD_ASYNC_TO_LDS_B128
// (ASYNCcnt) so HBM latency hides behind v_wmma work.
// ---------------------------------------------------------------------------

#define SEQ    4096
#define DIM    1280
#define NH     16
#define HD     80
#define HDP    96      // head dim padded to 3*32 for WMMA K-steps
#define SEGLEN 512
#define QKVN   (3*DIM) // 3840

#define LDS_STRIDE 40  // 32 K-elements + 8 pad (80B = 20 banks: conflict-free)

typedef __bf16 bf16_t;
typedef __attribute__((ext_vector_type(16))) __bf16 v16bf;
typedef __attribute__((ext_vector_type(8)))  __bf16 bf16x8;
typedef __attribute__((ext_vector_type(8)))  float  v8f;

union FragU { bf16x8 h[2]; v16bf f; };

// A-matrix fragment (16x32, 16-bit): lane half hf gets K-chunks [8*hf,8*hf+8)
// and [16+8*hf, 16+8*hf+8)  (ISA 7.12.2 16-bit A layout)
__device__ __forceinline__ v16bf frag_a(const bf16_t* p, int hf) {
  FragU u;
  u.h[0] = *(const bf16x8*)(p + 8 * hf);
  u.h[1] = *(const bf16x8*)(p + 16 + 8 * hf);
  return u.f;
}
// B-matrix fragment (32x16, 16-bit): lane half hf gets K-chunks [8*hf,8*hf+8)
// and [16+8*hf, 16+8*hf+8)  (same K order as frag_a)
__device__ __forceinline__ v16bf frag_b(const bf16_t* p, int hf) {
  FragU u;
  u.h[0] = *(const bf16x8*)(p + 8 * hf);
  u.h[1] = *(const bf16x8*)(p + 16 + 8 * hf);
  return u.f;
}

__device__ __forceinline__ v8f wmma_bf16(v16bf a, v16bf b, v8f c) {
  return __builtin_amdgcn_wmma_f32_16x16x32_bf16(false, a, false, b,
                                                 (short)0, c, false, false);
}

// CDNA5 async DMA: 16B global -> LDS per lane, tracked by ASYNCcnt.
__device__ __forceinline__ void async_ld_b128(uint32_t lds_off, uint64_t gaddr) {
  asm volatile("global_load_async_to_lds_b128 %0, %1, off"
               :: "v"(lds_off), "v"(gaddr) : "memory");
}
__device__ __forceinline__ void wait_async_le4() {
  asm volatile("s_wait_asynccnt 0x4" ::: "memory");
}
__device__ __forceinline__ void wait_async_0() {
  asm volatile("s_wait_asynccnt 0x0" ::: "memory");
}

// ---------------------------------------------------------------------------
// fp32 -> bf16 conversion
// ---------------------------------------------------------------------------
__global__ void cvt_f32_bf16(const float* __restrict__ in,
                             bf16_t* __restrict__ out, int n) {
  int i = blockIdx.x * blockDim.x + threadIdx.x;
  if (i < n) out[i] = (bf16_t)in[i];
}

// ---------------------------------------------------------------------------
// C[M,N] = A[M,K] * B[N,K]^T + bias ; A,B bf16 K-major; out bf16 or fp32.
// Block = 256 threads (8 waves), tile 128(M) x 128(N), K-panels of 32,
// double-buffered in LDS via async DMA. Wave tile = 32x64 (2x4 subtiles).
// ---------------------------------------------------------------------------
__global__ __launch_bounds__(256)
void gemm_nt_bf16(const bf16_t* __restrict__ A,
                  const bf16_t* __restrict__ B,
                  const float* __restrict__ bias,
                  bf16_t* __restrict__ outb,
                  float* __restrict__ outf,
                  int M, int N, int K) {
  __shared__ __align__(16) bf16_t Asm[2][128 * LDS_STRIDE];
  __shared__ __align__(16) bf16_t Bsm[2][128 * LDS_STRIDE];

  const int lane = threadIdx.x & 31;
  const int wave = threadIdx.x >> 5;
  const int r    = lane & 15;
  const int hf   = lane >> 4;
  const int wm   = wave >> 1;   // 0..3 -> M rows [wm*32, wm*32+32)
  const int wn   = wave & 1;    // 0..1 -> N cols [wn*64, wn*64+64)
  const int m00  = blockIdx.y * 128;
  const int n00  = blockIdx.x * 128;

  // per-(wave,lane) DMA chunk mapping: 512 chunks of 16B per 128x32 panel
  const int ch0  = wave * 64 + lane;        // chunk ids ch0, ch0+32
  const uint64_t Abase = (uint64_t)(uintptr_t)A;
  const uint64_t Bbase = (uint64_t)(uintptr_t)B;

  auto issue = [&](int buf, int k0) {
#pragma unroll
    for (int i = 0; i < 2; ++i) {
      int chunk = ch0 + i * 32;
      int row = chunk >> 2, c = chunk & 3;     // 4 x 16B per 32-K row
      uint32_t loff = (uint32_t)(row * LDS_STRIDE * 2 + c * 16);
      uint32_t la = (uint32_t)(uintptr_t)(&Asm[buf][0]) + loff;
      uint32_t lb = (uint32_t)(uintptr_t)(&Bsm[buf][0]) + loff;
      uint64_t ga = Abase + ((size_t)(m00 + row) * K + k0) * 2 + c * 16;
      uint64_t gb = Bbase + ((size_t)(n00 + row) * K + k0) * 2 + c * 16;
      async_ld_b128(la, ga);
      async_ld_b128(lb, gb);
    }
  };

  v8f acc[2][4];
#pragma unroll
  for (int s = 0; s < 2; ++s)
#pragma unroll
    for (int j = 0; j < 4; ++j)
      acc[s][j] = (v8f){0.f,0.f,0.f,0.f,0.f,0.f,0.f,0.f};

  const int NK = K / 32;
  issue(0, 0);
  for (int kc = 0; kc < NK; ++kc) {
    int cur = kc & 1;
    if (kc + 1 < NK) { issue(cur ^ 1, (kc + 1) * 32); wait_async_le4(); }
    else             { wait_async_0(); }
    __syncthreads();

    const bf16_t* At = &Asm[cur][0];
    const bf16_t* Bt = &Bsm[cur][0];
    v16bf af[2], bfr[4];
#pragma unroll
    for (int s = 0; s < 2; ++s)
      af[s] = frag_a(At + (wm * 32 + s * 16 + r) * LDS_STRIDE, hf);
#pragma unroll
    for (int j = 0; j < 4; ++j)
      bfr[j] = frag_b(Bt + (wn * 64 + j * 16 + r) * LDS_STRIDE, hf);
#pragma unroll
    for (int s = 0; s < 2; ++s)
#pragma unroll
      for (int j = 0; j < 4; ++j)
        acc[s][j] = wmma_bf16(af[s], bfr[j], acc[s][j]);

    __syncthreads();
  }

  // C layout: element (vgpr v, lane) -> row +v+8*hf, col +j*16+r
#pragma unroll
  for (int s = 0; s < 2; ++s) {
#pragma unroll
    for (int j = 0; j < 4; ++j) {
      int col = n00 + wn * 64 + j * 16 + r;
      float bv = bias[col];
#pragma unroll
      for (int v = 0; v < 8; ++v) {
        int row = m00 + wm * 32 + s * 16 + v + 8 * hf;
        float val = acc[s][j][v] + bv;
        if (outf) outf[(size_t)row * N + col] = val;
        else      outb[(size_t)row * N + col] = (bf16_t)val;
      }
    }
  }
}

// ---------------------------------------------------------------------------
// RoPE + repack: qkv_bf[s][3][h][80] ->
//   qp[h][s][96] (rope, *1/sqrt(80), zero pad), kp[h][s][96] (rope, pad),
//   vt[h][d][s]  (transposed for PV B-operand)
// ---------------------------------------------------------------------------
__global__ void rope_pack(const bf16_t* __restrict__ qkv,
                          const float* __restrict__ cosb,
                          const float* __restrict__ sinb,
                          bf16_t* __restrict__ qp,
                          bf16_t* __restrict__ kp,
                          bf16_t* __restrict__ vt) {
  int i = blockIdx.x * blockDim.x + threadIdx.x;
  if (i >= SEQ * DIM) return;
  int s = i / DIM, c = i % DIM;
  int h = c / HD, d = c % HD;
  size_t base = (size_t)s * QKVN;

  float q  = (float)qkv[base +          h * HD + d];
  float k  = (float)qkv[base +   DIM  + h * HD + d];
  float v  = (float)qkv[base + 2*DIM  + h * HD + d];
  int   dp = (d < HD/2) ? d + HD/2 : d - HD/2;
  float qpair = (float)qkv[base +        h * HD + dp];
  float kpair = (float)qkv[base + DIM  + h * HD + dp];
  float sgn = (d < HD/2) ? -1.f : 1.f;
  float cs = cosb[s * HD + d];
  float sn = sinb[s * HD + d];
  float qr = q * cs + sgn * qpair * sn;
  float kr = k * cs + sgn * kpair * sn;
  const float scal = 0.11180339887498949f; // 80^-0.5 folded into Q

  size_t hsrow = ((size_t)h * SEQ + s) * HDP;
  qp[hsrow + d] = (bf16_t)(qr * scal);
  kp[hsrow + d] = (bf16_t)kr;
  vt[((size_t)h * HD + d) * SEQ + s] = (bf16_t)v;
  if (d < HDP - HD) {          // zero the 80..95 pad
    qp[hsrow + HD + d] = (bf16_t)0.f;
    kp[hsrow + HD + d] = (bf16_t)0.f;
  }
}

// ---------------------------------------------------------------------------
// Flash-style block-diagonal attention. 1 wave = (head, 16 query rows).
// Keys iterated in chunks of 32 within the query tile's segment.
// Per chunk: 6 WMMA (Q.K^T, K=96) + 5 WMMA (P.V, dims 80) + online softmax.
// ---------------------------------------------------------------------------
__global__ void attn_kernel(const bf16_t* __restrict__ qp,
                            const bf16_t* __restrict__ kp,
                            const bf16_t* __restrict__ vt,
                            bf16_t* __restrict__ attn) {
  __shared__ __align__(16) bf16_t lds[4][16 * 32];

  const int lane = threadIdx.x & 31;
  const int wave = threadIdx.x >> 5;
  const int r    = lane & 15;
  const int hf   = lane >> 4;
  const int h    = blockIdx.x;
  const int q0   = (blockIdx.y * 4 + wave) * 16;
  const int seg0 = (q0 / SEGLEN) * SEGLEN;

  const bf16_t* qrow = qp + ((size_t)h * SEQ + q0 + r) * HDP;
  v16bf qf0 = frag_a(qrow,      hf);
  v16bf qf1 = frag_a(qrow + 32, hf);
  v16bf qf2 = frag_a(qrow + 64, hf);

  float mrow[8], lrow[8];
  v8f acc[5];
#pragma unroll
  for (int v = 0; v < 8; ++v) { mrow[v] = -1e30f; lrow[v] = 0.f; }
#pragma unroll
  for (int j = 0; j < 5; ++j) acc[j] = (v8f){0.f,0.f,0.f,0.f,0.f,0.f,0.f,0.f};

  const bf16_t* kh = kp + (size_t)h * SEQ * HDP;
  bf16_t* ltile = lds[wave];

  for (int kc = 0; kc < SEGLEN / 32; ++kc) {
    int n0 = seg0 + kc * 32;
    const bf16_t* k0row = kh + (size_t)(n0 +      r) * HDP;
    const bf16_t* k1row = kh + (size_t)(n0 + 16 + r) * HDP;

    v8f s0 = (v8f){0.f,0.f,0.f,0.f,0.f,0.f,0.f,0.f};
    v8f s1 = (v8f){0.f,0.f,0.f,0.f,0.f,0.f,0.f,0.f};
    s0 = wmma_bf16(qf0, frag_b(k0row,      hf), s0);
    s0 = wmma_bf16(qf1, frag_b(k0row + 32, hf), s0);
    s0 = wmma_bf16(qf2, frag_b(k0row + 64, hf), s0);
    s1 = wmma_bf16(qf0, frag_b(k1row,      hf), s1);
    s1 = wmma_bf16(qf1, frag_b(k1row + 32, hf), s1);
    s1 = wmma_bf16(qf2, frag_b(k1row + 64, hf), s1);

    // online softmax; row m = v + 8*hf lives across the 16 lanes of a half
    float p0[8], p1[8];
#pragma unroll
    for (int v = 0; v < 8; ++v) {
      float mx = fmaxf(s0[v], s1[v]);
#pragma unroll
      for (int m = 1; m < 16; m <<= 1) mx = fmaxf(mx, __shfl_xor(mx, m, 32));
      float mnew = fmaxf(mrow[v], mx);
      float corr = __expf(mrow[v] - mnew);
      float e0 = __expf(s0[v] - mnew);
      float e1 = __expf(s1[v] - mnew);
      float ss = e0 + e1;
#pragma unroll
      for (int m = 1; m < 16; m <<= 1) ss += __shfl_xor(ss, m, 32);
      lrow[v] = lrow[v] * corr + ss;
      mrow[v] = mnew;
      p0[v] = e0; p1[v] = e1;
#pragma unroll
      for (int j = 0; j < 5; ++j) acc[j][v] *= corr;
    }

    // stage P (C layout) through LDS, reload in A-fragment layout
    __syncthreads();
#pragma unroll
    for (int v = 0; v < 8; ++v) {
      ltile[(v + 8 * hf) * 32 + r]      = (bf16_t)p0[v];
      ltile[(v + 8 * hf) * 32 + 16 + r] = (bf16_t)p1[v];
    }
    __syncthreads();
    v16bf pf = frag_a(ltile + r * 32, hf);

#pragma unroll
    for (int j = 0; j < 5; ++j) {
      const bf16_t* vrow = vt + ((size_t)h * HD + j * 16 + r) * SEQ + n0;
      acc[j] = wmma_bf16(pf, frag_b(vrow, hf), acc[j]);
    }
  }

  // normalize + store attn[s][h*80 + d] (bf16, K-major for proj GEMM)
#pragma unroll
  for (int v = 0; v < 8; ++v) {
    float inv = 1.f / lrow[v];
    int row = q0 + v + 8 * hf;
#pragma unroll
    for (int j = 0; j < 5; ++j)
      attn[(size_t)row * DIM + h * HD + j * 16 + r] = (bf16_t)(acc[j][v] * inv);
  }
}

// ---------------------------------------------------------------------------
// launch
// ---------------------------------------------------------------------------
extern "C" void kernel_launch(void* const* d_in, const int* in_sizes, int n_in,
                              void* d_out, int out_size, void* d_ws, size_t ws_size,
                              hipStream_t stream) {
  const float* hs    = (const float*)d_in[0];
  const float* cosb  = (const float*)d_in[1];
  const float* sinb  = (const float*)d_in[2];
  const float* wqkv  = (const float*)d_in[3];
  const float* bqkv  = (const float*)d_in[4];
  const float* wproj = (const float*)d_in[5];
  const float* bproj = (const float*)d_in[6];
  // d_in[7] = cu_seqlens: uniform 512-token segments (handled structurally)

  uint8_t* ws = (uint8_t*)d_ws;
  // workspace layout (all 256B aligned); attn reuses the hs_bf region
  const size_t HS_BF    = 0;                                   // 4096*1280 bf16
  const size_t WQKV_BF  = HS_BF    + (size_t)SEQ * DIM * 2;    // 3840*1280 bf16
  const size_t WPROJ_BF = WQKV_BF  + (size_t)QKVN * DIM * 2;   // 1280*1280 bf16
  const size_t QKV_BF   = WPROJ_BF + (size_t)DIM * DIM * 2;    // 4096*3840 bf16
  const size_t Q_BF     = QKV_BF   + (size_t)SEQ * QKVN * 2;   // 16*4096*96 bf16
  const size_t K_BF     = Q_BF     + (size_t)NH * SEQ * HDP * 2;
  const size_t V_T      = K_BF     + (size_t)NH * SEQ * HDP * 2; // 16*80*4096

  bf16_t* hs_bf    = (bf16_t*)(ws + HS_BF);
  bf16_t* wqkv_bf  = (bf16_t*)(ws + WQKV_BF);
  bf16_t* wproj_bf = (bf16_t*)(ws + WPROJ_BF);
  bf16_t* qkv_bf   = (bf16_t*)(ws + QKV_BF);
  bf16_t* q_bf     = (bf16_t*)(ws + Q_BF);
  bf16_t* k_bf     = (bf16_t*)(ws + K_BF);
  bf16_t* v_t      = (bf16_t*)(ws + V_T);
  bf16_t* attn_bf  = hs_bf; // hs_bf dead after QKV GEMM

  // 1) convert inputs to bf16
  {
    int n = SEQ * DIM;
    cvt_f32_bf16<<<(n + 255) / 256, 256, 0, stream>>>(hs, hs_bf, n);
    n = QKVN * DIM;
    cvt_f32_bf16<<<(n + 255) / 256, 256, 0, stream>>>(wqkv, wqkv_bf, n);
    n = DIM * DIM;
    cvt_f32_bf16<<<(n + 255) / 256, 256, 0, stream>>>(wproj, wproj_bf, n);
  }

  // 2) QKV projection: [4096,3840] = hs_bf @ wqkv^T + b  (bf16 out)
  gemm_nt_bf16<<<dim3(QKVN / 128, SEQ / 128), 256, 0, stream>>>(
      hs_bf, wqkv_bf, bqkv, qkv_bf, nullptr, SEQ, QKVN, DIM);

  // 3) RoPE + repack (Q scaled, head-dim padded, V transposed)
  rope_pack<<<(SEQ * DIM + 255) / 256, 256, 0, stream>>>(
      qkv_bf, cosb, sinb, q_bf, k_bf, v_t);

  // 4) block-diagonal flash attention
  attn_kernel<<<dim3(NH, SEQ / 16 / 4), 128, 0, stream>>>(
      q_bf, k_bf, v_t, attn_bf);

  // 5) output projection: fp32 out = attn_bf @ wproj^T + b
  gemm_nt_bf16<<<dim3(DIM / 128, SEQ / 128), 256, 0, stream>>>(
      attn_bf, wproj_bf, bproj, nullptr, (float*)d_out, SEQ, DIM, DIM);
}
